// VectorQuantizer_77721728188769
// MI455X (gfx1250) — matrix operand order repaired and checked by reading, not compiled, not run on hardware
//
#include <hip/hip_runtime.h>

typedef unsigned short u16;
typedef __attribute__((ext_vector_type(16))) __bf16 v16bf;
typedef __attribute__((ext_vector_type(16))) u16    v16u;
typedef __attribute__((ext_vector_type(8)))  u16    v8u;
typedef __attribute__((ext_vector_type(8)))  float  v8f;
typedef __attribute__((ext_vector_type(4)))  int    v4i;

union BfFrag { v16u u; v16bf b; };

__device__ __forceinline__ u16 f32_to_bf16_rne(float f) {
  unsigned int u = __float_as_uint(f);
  unsigned int r = u + 0x7FFFu + ((u >> 16) & 1u);
  return (u16)(r >> 16);
}
__device__ __forceinline__ float bf16_to_f32(u16 h) {
  return __uint_as_float(((unsigned int)h) << 16);
}

// Async global->LDS 16B copy (gfx1250 GLOBAL_LOAD_ASYNC_TO_LDS_B128, ASYNCcnt).
// The builtin wants typed int4* pointers: as(1) for the global source, as(3)
// for the LDS destination. Generic AMDGPU pointers keep the LDS offset in the
// low 32 bits, so the integer round-trip cast produces valid pointers.
typedef __attribute__((address_space(1))) v4i* g_v4i_p;
typedef __attribute__((address_space(3))) v4i* l_v4i_p;

__device__ __forceinline__ void async_copy_b128(const void* g, void* l) {
  __builtin_amdgcn_global_load_async_to_lds_b128(
      (g_v4i_p)(uintptr_t)g,
      (l_v4i_p)(unsigned int)(uintptr_t)l,
      0, 0);
}
__device__ __forceinline__ void wait_async_all() {
  asm volatile("s_wait_asynccnt 0x0" ::: "memory");
}

// ---------------------------------------------------------------------------
// prep_embed: embeddings (128 x 1024, d-major) ->
//   ETf : code-major f32   [1024][128]  (exact values for final gather)
//   ETh : code-major bf16 hi-part [1024][128]
//   ETl : code-major bf16 lo-part [1024][128]
// ---------------------------------------------------------------------------
__global__ __launch_bounds__(256) void prep_embed(
    const float* __restrict__ emb, float* __restrict__ ETf,
    u16* __restrict__ ETh, u16* __restrict__ ETl)
{
  int t = blockIdx.x * 256 + threadIdx.x;   // 0 .. 131071
  int i = t >> 10;                          // dim   0..127
  int j = t & 1023;                         // code  0..1023
  float e = emb[t];                         // emb[i*1024 + j]
  u16 h = f32_to_bf16_rne(e);
  ETf[j * 128 + i] = e;
  ETh[j * 128 + i] = h;
  ETl[j * 128 + i] = f32_to_bf16_rne(e - bf16_to_f32(h));
}

// ---------------------------------------------------------------------------
// prep_norm: enorm[j] = sum_i emb[i][j]^2   (coalesced across j)
// ---------------------------------------------------------------------------
__global__ __launch_bounds__(256) void prep_norm(
    const float* __restrict__ emb, float* __restrict__ enorm)
{
  int j = blockIdx.x * 256 + threadIdx.x;   // 0..1023
  float s = 0.f;
  #pragma unroll 8
  for (int i = 0; i < 128; ++i) {
    float v = emb[i * 1024 + j];
    s = fmaf(v, v, s);
  }
  enorm[j] = s;
}

// ---------------------------------------------------------------------------
// vq_main: 256 threads = 8 waves, each wave owns 16 rows (block = 128 rows).
// Fused GEMM(argmin) with bf16 split-precision WMMA, double-buffered async
// global->LDS staging of the code tiles, exact f32 gather at the end.
// ---------------------------------------------------------------------------
__global__ __launch_bounds__(256) void vq_main(
    const float* __restrict__ x,  const float* __restrict__ ETf,
    const u16* __restrict__ ETh,  const u16* __restrict__ ETl,
    const float* __restrict__ enorm, float* __restrict__ out)
{
  __shared__ float xs[128 * 132];        // 128 rows, stride 132 (bank spread)
  __shared__ u16   eHi[2][16 * 136];     // double-buffered 16-code bf16 tiles
  __shared__ u16   eLo[2][16 * 136];
  __shared__ float enLDS[1024];

  const int tid  = threadIdx.x;
  const int lane = tid & 31;
  const int wave = tid >> 5;
  const int R0   = blockIdx.x * 128;

  // staging geometry for the e-tiles: one uint4 per thread per array
  const int sr  = tid >> 4;              // code row 0..15
  const int sc8 = (tid & 15) << 3;       // k-chunk of 8 bf16 (16B)
  const int ldsOff = sr * 136 + sc8;

  // ---- kick off async staging of code tile 0 into buffer 0 ----------------
  async_copy_b128(&ETh[(size_t)sr * 128 + sc8], &eHi[0][ldsOff]);
  async_copy_b128(&ETl[(size_t)sr * 128 + sc8], &eLo[0][ldsOff]);

  // ---- stage this block's 128 x rows into LDS (coalesced b128) + enorm ----
  for (int p = tid; p < 4096; p += 256) {
    int r = p >> 5, c = (p & 31) << 2;
    *(float4*)&xs[r * 132 + c] = *(const float4*)&x[(size_t)(R0 + r) * 128 + c];
  }
  for (int p = tid; p < 1024; p += 256) enLDS[p] = enorm[p];
  wait_async_all();
  __syncthreads();

  // ---- build A fragments once (16x32 bf16 layout, hi/lo split) ------------
  // lane<16: row m=lane, elems = K{0..7, 16..23}; lane>=16: K{8..15, 24..31}
  const int m    = lane & 15;
  const int koff = (lane >> 4) << 3;          // 0 or 8
  const float* xr = &xs[(wave * 16 + m) * 132];
  BfFrag Ah[4], Al[4];
  #pragma unroll
  for (int kt = 0; kt < 4; ++kt) {
    int k0 = kt * 32 + koff;
    #pragma unroll
    for (int c = 0; c < 8; ++c) {
      float f0 = xr[k0 + c];
      float f1 = xr[k0 + 16 + c];
      u16 h0 = f32_to_bf16_rne(f0);
      u16 h1 = f32_to_bf16_rne(f1);
      Ah[kt].u[c]     = h0;
      Ah[kt].u[c + 8] = h1;
      Al[kt].u[c]     = f32_to_bf16_rne(f0 - bf16_to_f32(h0));
      Al[kt].u[c + 8] = f32_to_bf16_rne(f1 - bf16_to_f32(h1));
    }
  }

  float minv[8];
  int   mini[8];
  #pragma unroll
  for (int r = 0; r < 8; ++r) { minv[r] = 3.4e38f; mini[r] = 0; }

  const int jlane = lane & 15;                // B column = C column
  const int kboff = (lane >> 4) << 3;         // B rows: {0..7,16..23} or {8..15,24..31}

  for (int jt = 0; jt < 64; ++jt) {
    const int buf = jt & 1;

    // prefetch next tile into the other buffer while we compute (uniform br.)
    if (jt < 63) {
      async_copy_b128(&ETh[(size_t)((jt + 1) * 16 + sr) * 128 + sc8],
                      &eHi[buf ^ 1][ldsOff]);
      async_copy_b128(&ETl[(size_t)((jt + 1) * 16 + sr) * 128 + sc8],
                      &eLo[buf ^ 1][ldsOff]);
    }

    v8f acc0 = {};   // hi*hi
    v8f acc1 = {};   // hi*lo + lo*hi  (error-compensation terms)
    const u16* ehr = &eHi[buf][jlane * 136];
    const u16* elr = &eLo[buf][jlane * 136];
    #pragma unroll
    for (int kt = 0; kt < 4; ++kt) {
      int kb = kt * 32 + kboff;
      v8u bh0 = *(const v8u*)&ehr[kb];
      v8u bh1 = *(const v8u*)&ehr[kb + 16];
      v8u bl0 = *(const v8u*)&elr[kb];
      v8u bl1 = *(const v8u*)&elr[kb + 16];
      BfFrag Bh, Bl;
      Bh.u = __builtin_shufflevector(bh0, bh1, 0,1,2,3,4,5,6,7,8,9,10,11,12,13,14,15);
      Bl.u = __builtin_shufflevector(bl0, bl1, 0,1,2,3,4,5,6,7,8,9,10,11,12,13,14,15);
      acc0 = __builtin_amdgcn_wmma_f32_16x16x32_bf16(false, Ah[kt].b, false, Bh.b,
                                                     (short)0, acc0, false, false);
      acc1 = __builtin_amdgcn_wmma_f32_16x16x32_bf16(false, Ah[kt].b, false, Bl.b,
                                                     (short)0, acc1, false, false);
      acc1 = __builtin_amdgcn_wmma_f32_16x16x32_bf16(false, Al[kt].b, false, Bh.b,
                                                     (short)0, acc1, false, false);
    }

    // dist = ||e||^2 - 2*S  (||f||^2 constant per row: drop it)
    int   jcode = jt * 16 + jlane;
    float en    = enLDS[jcode];
    #pragma unroll
    for (int r = 0; r < 8; ++r) {
      float d = en - 2.0f * (acc0[r] + acc1[r]);
      if (d < minv[r]) { minv[r] = d; mini[r] = jcode; }   // jcode monotone
    }

    // publish the prefetched buffer to the whole workgroup
    wait_async_all();
    __syncthreads();
  }

  // ---- cross-lane argmin within each 16-lane half (C layout: col = l%16) --
  #pragma unroll
  for (int r = 0; r < 8; ++r) {
    #pragma unroll
    for (int off = 8; off >= 1; off >>= 1) {
      float ov = __shfl_xor(minv[r], off, 32);
      int   oi = __shfl_xor(mini[r], off, 32);
      if (ov < minv[r] || (ov == minv[r] && oi < mini[r])) {
        minv[r] = ov; mini[r] = oi;
      }
    }
  }

  // ---- gather: whole wave copies each of its 16 rows (512B contiguous) ----
  const int rowBase = R0 + wave * 16;
  #pragma unroll
  for (int mm = 0; mm < 16; ++mm) {
    int idx = __shfl(mini[mm & 7], (mm >> 3) << 4, 32);  // row mm lives in half mm/8
    const float4* src = (const float4*)&ETf[(size_t)idx * 128];
    float4*       dst = (float4*)&out[(size_t)(rowBase + mm) * 128];
    dst[lane] = src[lane];
  }
}

// ---------------------------------------------------------------------------
extern "C" void kernel_launch(void* const* d_in, const int* in_sizes, int n_in,
                              void* d_out, int out_size, void* d_ws, size_t ws_size,
                              hipStream_t stream) {
  const float* x   = (const float*)d_in[0];   // (64,32,32,128) f32
  const float* emb = (const float*)d_in[1];   // (128,1024) f32
  float* out = (float*)d_out;

  char* ws = (char*)d_ws;
  float* ETf   = (float*)(ws);                //  512 KB: code-major f32
  u16*   ETh   = (u16*)  (ws + 524288);       //  256 KB: bf16 hi
  u16*   ETl   = (u16*)  (ws + 786432);       //  256 KB: bf16 lo
  float* enorm = (float*)(ws + 1048576);      //    4 KB: ||e_j||^2

  prep_embed<<<512, 256, 0, stream>>>(emb, ETf, ETh, ETl);
  prep_norm <<<4,   256, 0, stream>>>(emb, enorm);

  int nrows = in_sizes[0] / 128;              // 65536
  vq_main<<<nrows / 128, 256, 0, stream>>>(x, ETf, ETh, ETl, enorm, out);
}
